// PatchCoreEffB3_45638322487406
// MI455X (gfx1250) — matrix operand order repaired and checked by reading, not compiled, not run on hardware
//
#include <hip/hip_runtime.h>

typedef __attribute__((ext_vector_type(16))) __bf16 v16bf;
typedef __attribute__((ext_vector_type(8)))  __bf16 v8bf;
typedef __attribute__((ext_vector_type(8)))  float  v8f;

#define C_DIM 176
#define KPAD  192
#define NKT   6      // K tiles of 32 -> 192
#define SPLITS 16

// ---------- helpers ----------
__device__ __forceinline__ unsigned short f32_to_bf16(float f) {
  unsigned u = __builtin_bit_cast(unsigned, f);
  u += 0x7FFFu + ((u >> 16) & 1u);          // round-to-nearest-even
  return (unsigned short)(u >> 16);
}

// order-preserving float <-> uint mapping (for integer atomicMax on floats)
__device__ __forceinline__ unsigned enc_f32(float f) {
  unsigned u = __builtin_bit_cast(unsigned, f);
  return (u & 0x80000000u) ? ~u : (u | 0x80000000u);
}
__device__ __forceinline__ float dec_f32(unsigned u) {
  unsigned b = (u & 0x80000000u) ? (u ^ 0x80000000u) : ~u;
  return __builtin_bit_cast(float, b);
}

// single-instruction max (avoids fmaxf's sNaN-canonicalization self-maxes)
__device__ __forceinline__ float max_num(float a, float b) {
  float d;
  asm("v_max_num_f32 %0, %1, %2" : "=v"(d) : "v"(a), "v"(b));
  return d;
}

// ---------- kernel 1: row L2-normalize + fp32 -> bf16, pad K 176 -> 192 ----------
__global__ void __launch_bounds__(256) normalize_bf16_kernel(const float* __restrict__ src,
                                                             unsigned short* __restrict__ dst,
                                                             int rows) {
  int wave = threadIdx.x >> 5;
  int lane = threadIdx.x & 31;
  int row  = blockIdx.x * 8 + wave;          // 8 waves per block, 1 row per wave
  if (row >= rows) return;

  const float* s = src + (size_t)row * C_DIM;
  float vals[6];
  float ss = 0.f;
  int cnt = 0;
  for (int c = lane; c < C_DIM; c += 32) { float v = s[c]; vals[cnt++] = v; ss += v * v; }
  #pragma unroll
  for (int off = 16; off >= 1; off >>= 1) ss += __shfl_xor(ss, off, 32);
  float scale = 1.0f / (sqrtf(ss) + 1e-6f);

  unsigned short* d = dst + (size_t)row * KPAD;
  cnt = 0;
  for (int c = lane; c < C_DIM; c += 32) d[c] = f32_to_bf16(vals[cnt++] * scale);
  for (int c = C_DIM + lane; c < KPAD; c += 32) d[c] = 0;   // zero pad (bf16 0.0)
}

// ---------- kernel 2: init per-row encoded max ----------
__global__ void __launch_bounds__(256) init_max_kernel(unsigned* __restrict__ maxenc, int n) {
  int i = blockIdx.x * blockDim.x + threadIdx.x;
  if (i < n) maxenc[i] = 0u;                 // below any real encoded sim
}

// ---------- kernel 3: WMMA similarity + running max over memory bank ----------
// grid.x = N/32 (each block/wave covers TWO 16-row A tiles -> 2x B reuse),
// grid.y = SPLITS (slices of the M dimension). 8 waves / block; each wave
// strides over m-tiles with a double-buffered B pipeline and two independent
// accumulator chains. nIter (tiles per wave) must be even >= 2.
#define LOADB(buf)                                                              \
  {                                                                             \
    _Pragma("unroll")                                                           \
    for (int kk = 0; kk < NKT; ++kk) buf[kk] = *(const v16bf*)(bptr + kk * 32); \
    bptr += stepElems;                                                          \
  }

#define COMPUTE(buf)                                                            \
  {                                                                             \
    v8f acc0 = {0.f, 0.f, 0.f, 0.f, 0.f, 0.f, 0.f, 0.f};                        \
    v8f acc1 = {0.f, 0.f, 0.f, 0.f, 0.f, 0.f, 0.f, 0.f};                        \
    _Pragma("unroll")                                                           \
    for (int kk = 0; kk < NKT; ++kk) {                                          \
      acc0 = __builtin_amdgcn_wmma_f32_16x16x32_bf16(                           \
               false, afragA[kk], false, buf[kk], (short)0, acc0, false, false);\
      acc1 = __builtin_amdgcn_wmma_f32_16x16x32_bf16(                           \
               false, afragB[kk], false, buf[kk], (short)0, acc1, false, false);\
    }                                                                           \
    _Pragma("unroll")                                                           \
    for (int r = 0; r < 8; ++r) {                                               \
      vmax0[r] = max_num(vmax0[r], acc0[r]);                                    \
      vmax1[r] = max_num(vmax1[r], acc1[r]);                                    \
    }                                                                           \
  }

__global__ void __launch_bounds__(256) simmax_kernel(const unsigned short* __restrict__ pA,
                                                     const unsigned short* __restrict__ pM,
                                                     unsigned* __restrict__ maxenc,
                                                     int tilesPerSplit, int nIter) {
  int tid  = threadIdx.x;
  int wave = __builtin_amdgcn_readfirstlane(tid >> 5);  // SGPR -> scalar loop, EXEC untouched
  int lane = tid & 31;
  int l    = lane & 15;
  int hi   = lane >> 4;
  int rowTile = blockIdx.x << 5;            // 32 patch rows per block

  // A fragments (16x32 bf16 each), contiguous-K order (same as B):
  //   lanes 0-15 : row l, K = k0+0..15 contiguous
  //   lanes 16-31: row l, K = k0+16..31 contiguous
  const unsigned short* arow0 = pA + (size_t)(rowTile + l) * KPAD;
  const unsigned short* arow1 = arow0 + (size_t)16 * KPAD;
  v16bf afragA[NKT], afragB[NKT];
  #pragma unroll
  for (int kk = 0; kk < NKT; ++kk) {
    int k0 = kk * 32;
    v8bf lo0 = *(const v8bf*)(arow0 + k0 + hi * 16);
    v8bf hv0 = *(const v8bf*)(arow0 + k0 + hi * 16 + 8);
    afragA[kk] = __builtin_shufflevector(lo0, hv0,
                   0,1,2,3,4,5,6,7,8,9,10,11,12,13,14,15);
    v8bf lo1 = *(const v8bf*)(arow1 + k0 + hi * 16);
    v8bf hv1 = *(const v8bf*)(arow1 + k0 + hi * 16 + 8);
    afragB[kk] = __builtin_shufflevector(lo1, hv1,
                   0,1,2,3,4,5,6,7,8,9,10,11,12,13,14,15);
  }

  float vmax0[8], vmax1[8];
  #pragma unroll
  for (int r = 0; r < 8; ++r) { vmax0[r] = -3.0e38f; vmax1[r] = -3.0e38f; }

  // B fragment (32x16 bf16): lane n<16 = column n, K k0+0..15 contiguous;
  //                          lane n>=16 = column n-16, K k0+16..31 contiguous.
  int t0 = blockIdx.y * tilesPerSplit + wave;
  const unsigned short* bptr = pM + ((size_t)t0 * 16 + l) * KPAD + hi * 16;
  const size_t stepElems = (size_t)8 * 16 * KPAD;   // advance 8 m-tiles per step

  v16bf b0[NKT], b1[NKT];
  LOADB(b0);                                // prologue: tile (wave)
  for (int it = 0; it < nIter; it += 2) {   // uniform scalar loop
    LOADB(b1);                              // prefetch tile it+1
    COMPUTE(b0);
    if (it + 2 < nIter) LOADB(b0);          // prefetch tile it+2 (uniform branch)
    COMPUTE(b1);
  }

  // reduce max across the 16 columns held by each half-wave group
  #pragma unroll
  for (int r = 0; r < 8; ++r) {
    float v0 = vmax0[r];
    float v1 = vmax1[r];
    #pragma unroll
    for (int off = 8; off >= 1; off >>= 1) {
      v0 = max_num(v0, __shfl_xor(v0, off, 16));
      v1 = max_num(v1, __shfl_xor(v1, off, 16));
    }
    vmax0[r] = v0;
    vmax1[r] = v1;
  }
  if (l == 0) {
    int rowBase = rowTile + hi * 8;   // lanes 0-15 hold rows 0..7; 16-31 hold 8..15
    #pragma unroll
    for (int r = 0; r < 8; ++r) {
      atomicMax(&maxenc[rowBase + r],      enc_f32(vmax0[r]));
      atomicMax(&maxenc[rowBase + 16 + r], enc_f32(vmax1[r]));
    }
  }
}

// ---------- kernel 4: scores + top-k mean ----------
__global__ void __launch_bounds__(256) finalize_kernel(const unsigned* __restrict__ maxenc,
                                                       float* __restrict__ out,
                                                       int N, const int* __restrict__ topk_ptr) {
  __shared__ float sc[4096];
  __shared__ float rmax[256];
  __shared__ int   ridx[256];
  int tid = threadIdx.x;
  int topk = *topk_ptr;

  for (int i = tid; i < N; i += 256) {
    float score = 1.0f - dec_f32(maxenc[i]);
    sc[i]  = score;
    out[i] = score;
  }
  __syncthreads();

  float sum = 0.f;
  for (int it = 0; it < topk; ++it) {
    float best = -3.0e38f; int bidx = 0;
    for (int i = tid; i < N; i += 256) {
      float v = sc[i];
      if (v > best) { best = v; bidx = i; }
    }
    rmax[tid] = best; ridx[tid] = bidx;
    __syncthreads();
    for (int s = 128; s > 0; s >>= 1) {
      if (tid < s && rmax[tid + s] > rmax[tid]) {
        rmax[tid] = rmax[tid + s]; ridx[tid] = ridx[tid + s];
      }
      __syncthreads();
    }
    if (tid == 0) { sum += rmax[0]; sc[ridx[0]] = -3.0e38f; }
    __syncthreads();
  }
  if (tid == 0) out[N] = sum / (float)topk;
}

// ---------- host launcher ----------
extern "C" void kernel_launch(void* const* d_in, const int* in_sizes, int n_in,
                              void* d_out, int out_size, void* d_ws, size_t ws_size,
                              hipStream_t stream) {
  const float* patch = (const float*)d_in[0];
  const float* mem   = (const float*)d_in[1];
  const int*   topk  = (const int*)d_in[2];
  int N = in_sizes[0] / C_DIM;   // 4096
  int M = in_sizes[1] / C_DIM;   // 131072
  float* out = (float*)d_out;

  // workspace layout: [ pM bf16 M*192 | pA bf16 N*192 | maxenc u32 N ]
  char* ws = (char*)d_ws;
  unsigned short* pM = (unsigned short*)ws;
  size_t mBytes = (size_t)M * KPAD * sizeof(unsigned short);
  unsigned short* pA = (unsigned short*)(ws + mBytes);
  size_t aBytes = (size_t)N * KPAD * sizeof(unsigned short);
  unsigned* maxenc = (unsigned*)(ws + mBytes + aBytes);

  normalize_bf16_kernel<<<(M + 7) / 8, 256, 0, stream>>>(mem,   pM, M);
  normalize_bf16_kernel<<<(N + 7) / 8, 256, 0, stream>>>(patch, pA, N);
  init_max_kernel<<<(N + 255) / 256, 256, 0, stream>>>(maxenc, N);

  int tilesPerSplit = (M / 16) / SPLITS;   // 512
  int nIter = tilesPerSplit / 8;           // 64 tiles per wave (even)
  dim3 grid(N / 32, SPLITS);               // two 16-row A tiles per block
  simmax_kernel<<<grid, 256, 0, stream>>>(pA, pM, maxenc, tilesPerSplit, nIter);

  finalize_kernel<<<1, 256, 0, stream>>>(maxenc, out, N, topk);
}
